// QuantizedLinear_58007828300507
// MI455X (gfx1250) — compile-verified, matrix operand order checked by reading, not run on hardware
//
#include <hip/hip_runtime.h>
#include <hip/hip_bf16.h>
#include <stdint.h>

typedef __attribute__((ext_vector_type(16))) __bf16  v16bf;
typedef __attribute__((ext_vector_type(8)))  __bf16  v8bf;
typedef __attribute__((ext_vector_type(4)))  __bf16  v4bf;
typedef __attribute__((ext_vector_type(8)))  float   v8f;

#define M_DIM   8192     // BATCH * SEQ
#define N_DIM   4096     // OUT_FEATURES
#define K_DIM   4096     // IN_FEATURES
#define BM      128
#define BN      128
#define BK      32
#define LDSROW  40       // 32 bf16 + 8 pad -> 80B row stride (multiple of 16B)
#define NKT     (K_DIM / BK)

// Gather one 16x32 bf16 fragment (A-layout) from an LDS row pointer.
// Lane L: row = L%16 ; K-chunks at element offsets (L>=16? 8:0) and +16.
__device__ __forceinline__ v16bf lds_frag(const __bf16* p) {
  v8bf lo = *(const v8bf*)p;
  v8bf hi = *(const v8bf*)(p + 16);
  return __builtin_shufflevector(lo, hi, 0,1,2,3,4,5,6,7,8,9,10,11,12,13,14,15);
}

__global__ __launch_bounds__(256)
void qlinear_wmma_bf16(const float* __restrict__ x,
                       const int8_t* __restrict__ w,
                       const float* __restrict__ scale,
                       const float* __restrict__ bias,
                       float* __restrict__ out) {
  __shared__ __bf16 Als[2][BM][LDSROW];
  __shared__ __bf16 Bls[2][BN][LDSROW];

  const int tid  = threadIdx.x;
  const int gm   = blockIdx.y * BM;
  const int gn   = blockIdx.x * BN;

  const int wave = tid >> 5;
  const int lane = tid & 31;
  const int wm   = wave & 3;      // 4 wave-rows  * 32 M
  const int wn   = wave >> 2;     // 2 wave-cols  * 64 N
  const int lr   = lane & 15;     // row within 16x16 tile
  const int lh   = lane >> 4;     // K-half select

  // ---- global -> convert -> LDS staging for K-tile kt into buffer b
  auto stage = [&](int b, int kt) {
    const int k0 = kt * BK;
    // A: x tile [BM x BK] fp32 -> bf16 (truncate)
    #pragma unroll
    for (int i = 0; i < 4; ++i) {
      const int idx = i * 256 + tid;
      const int row = idx >> 3;
      const int c4  = idx & 7;
      const float4 v = *(const float4*)(x + (size_t)(gm + row) * K_DIM + k0 + c4 * 4);
      v4bf pk;
      pk[0] = (__bf16)v.x; pk[1] = (__bf16)v.y;
      pk[2] = (__bf16)v.z; pk[3] = (__bf16)v.w;
      *(v4bf*)&Als[b][row][c4 * 4] = pk;
    }
    // B: weight tile [BN x BK] int8 -> bf16 (exact; scale deferred to epilogue)
    {
      const int row  = tid >> 1;
      const int half = tid & 1;
      const int4 pk = *(const int4*)(w + (size_t)(gn + row) * K_DIM + k0 + half * 16);
      const int wd[4] = {pk.x, pk.y, pk.z, pk.w};
      v8bf b0, b1;
      #pragma unroll
      for (int q = 0; q < 4; ++q) {
        #pragma unroll
        for (int t = 0; t < 4; ++t) {
          const __bf16 e = (__bf16)(float)(int8_t)(wd[q] >> (8 * t));
          if (q < 2) b0[q * 4 + t] = e; else b1[(q - 2) * 4 + t] = e;
        }
      }
      *(v8bf*)&Bls[b][row][half * 16]     = b0;
      *(v8bf*)&Bls[b][row][half * 16 + 8] = b1;
    }
  };

  v8f acc[2][4];
  const v8f vzero = {0.f, 0.f, 0.f, 0.f, 0.f, 0.f, 0.f, 0.f};
  #pragma unroll
  for (int i = 0; i < 2; ++i)
    #pragma unroll
    for (int j = 0; j < 4; ++j)
      acc[i][j] = vzero;

  int buf = 0;
  stage(0, 0);

  for (int kt = 0; kt < NKT; ++kt) {
    __syncthreads();
    if (kt + 1 < NKT) stage(buf ^ 1, kt + 1);
    if (kt + 2 < NKT) {
      const int k2 = (kt + 2) * BK;
      __builtin_prefetch(x + (size_t)(gm + (tid >> 3)) * K_DIM + k2, 0, 1);
      __builtin_prefetch(w + (size_t)(gn + (tid >> 1)) * K_DIM + k2, 0, 1);
    }

    // Load fragments from LDS (A-layout for both; weight[N,K] == B^T)
    v16bf afrag[2], bfrag[4];
    #pragma unroll
    for (int i = 0; i < 2; ++i)
      afrag[i] = lds_frag(&Als[buf][wm * 32 + i * 16 + lr][lh * 8]);
    #pragma unroll
    for (int j = 0; j < 4; ++j)
      bfrag[j] = lds_frag(&Bls[buf][wn * 64 + j * 16 + lr][lh * 8]);

    #pragma unroll
    for (int i = 0; i < 2; ++i)
      #pragma unroll
      for (int j = 0; j < 4; ++j)
        acc[i][j] = __builtin_amdgcn_wmma_f32_16x16x32_bf16(
            /*neg_a=*/false, afrag[i],
            /*neg_b=*/false, bfrag[j],
            /*c_mod=*/(short)0, acc[i][j],
            /*reuse_a=*/false, /*reuse_b=*/false);

    buf ^= 1;
  }

  // ---- epilogue: D layout lane->N (lr), VGPR v -> M = v + 8*lh
  #pragma unroll
  for (int j = 0; j < 4; ++j) {
    const int   n0 = gn + wn * 64 + j * 16 + lr;
    const float s  = scale[n0];
    const float bb = bias[n0];
    #pragma unroll
    for (int i = 0; i < 2; ++i) {
      const int m0 = gm + wm * 32 + i * 16 + lh * 8;
      #pragma unroll
      for (int v = 0; v < 8; ++v)
        out[(size_t)(m0 + v) * N_DIM + n0] = acc[i][j][v] * s + bb;
    }
  }
}

extern "C" void kernel_launch(void* const* d_in, const int* in_sizes, int n_in,
                              void* d_out, int out_size, void* d_ws, size_t ws_size,
                              hipStream_t stream) {
  const float*  x     = (const float*)d_in[0];
  const int8_t* w     = (const int8_t*)d_in[1];
  const float*  scale = (const float*)d_in[2];
  const float*  bias  = (const float*)d_in[3];
  float*        out   = (float*)d_out;

  dim3 grid(N_DIM / BN, M_DIM / BM);   // 32 x 64 = 2048 workgroups
  qlinear_wmma_bf16<<<grid, dim3(256), 0, stream>>>(x, w, scale, bias, out);
}
